// MultiHeadAttention_31009663877820
// MI455X (gfx1250) — matrix operand order repaired and checked by reading, not compiled, not run on hardware
//
#include <hip/hip_runtime.h>

typedef __bf16 bf16;
typedef __attribute__((ext_vector_type(16))) __bf16 v16bf;
typedef __attribute__((ext_vector_type(8)))  __bf16 v8bf;
typedef __attribute__((ext_vector_type(8)))  float  v8f;

static constexpr int B_  = 2;
static constexpr int S_  = 2048;
static constexpr int D_  = 1024;
static constexpr int H_  = 16;
static constexpr int DH_ = 64;

union V16cvt { v16bf v; v8bf p[2]; };

// ---------------------------------------------------------------------------
// WMMA helpers (CDNA5 wave32, 16x16x32 bf16 -> f32)
// ---------------------------------------------------------------------------
__device__ inline v8f wmma_bf16(v16bf a, v16bf b, v8f c) {
    return __builtin_amdgcn_wmma_f32_16x16x32_bf16(
        /*neg_a=*/false, a, /*neg_b=*/false, b,
        /*c_mod=*/(short)0, c, /*reuse_a=*/false, /*reuse_b=*/false);
}

// A fragment: 16x32 tile (MxK), row-major source with leading dim lda.
// ISA layout: lane group 0 holds K = base+[0..7] and base+[16..23];
//             lane group 1 holds K = base+[8..15] and base+[24..31].
__device__ inline v16bf load_a_frag(const bf16* base, int lda, int row, int k,
                                    int lrow, int lgrp) {
    const bf16* r = base + (size_t)(row + lrow) * lda + k + lgrp * 8;
    V16cvt t;
    t.p[0] = *(const v8bf*)(r);
    t.p[1] = *(const v8bf*)(r + 16);
    return t.v;
}

// B fragment: 32x16 tile (KxN), from TRANSPOSED storage BT[n][k] with lead ldb.
// Lane group 0 holds K = base+[0..7] and base+[16..23]; group 1 K = base+[8..15] and base+[24..31].
__device__ inline v16bf load_b_frag(const bf16* baseT, int ldb, int col, int k,
                                    int lrow, int lgrp) {
    const bf16* r = baseT + (size_t)(col + lrow) * ldb + k + lgrp * 8;
    V16cvt t;
    t.p[0] = *(const v8bf*)(r);
    t.p[1] = *(const v8bf*)(r + 16);
    return t.v;
}

// ---------------------------------------------------------------------------
// Conversion kernels
// ---------------------------------------------------------------------------
__global__ void cvt_f32_bf16(const float* __restrict__ in, bf16* __restrict__ out, int n) {
    for (int i = blockIdx.x * blockDim.x + threadIdx.x; i < n;
         i += gridDim.x * blockDim.x)
        out[i] = (bf16)in[i];
}

// Wq [H][D][DH] -> WT [H][DH][D] (bf16)
__global__ void cvt_w_trans(const float* __restrict__ W, bf16* __restrict__ WT) {
    int i = blockIdx.x * blockDim.x + threadIdx.x;
    int total = H_ * DH_ * D_;
    if (i >= total) return;
    int d = i % D_;
    int r = i / D_;
    int e = r % DH_;
    int h = r / DH_;
    WT[i] = (bf16)W[((size_t)h * D_ + d) * DH_ + e];
}

// Wo [D][D] -> WoT [N][K] (bf16)
__global__ void cvt_wo_trans(const float* __restrict__ Wo, bf16* __restrict__ WoT) {
    int i = blockIdx.x * blockDim.x + threadIdx.x;
    if (i >= D_ * D_) return;
    int k = i % D_;
    int n = i / D_;
    WoT[i] = (bf16)Wo[(size_t)k * D_ + n];
}

// ---------------------------------------------------------------------------
// Per-head projection GEMM: each wave computes a 16 x DH (16x64) slab.
// Software-pipelined: k+32 fragments are loaded into fresh registers before
// the k fragments are consumed, so loads overlap WMMA issue.
// trans==0: out row-major [S][DH]; trans==1: out transposed [DH][S] (for V).
// ---------------------------------------------------------------------------
__global__ __launch_bounds__(32) void proj_kernel(const bf16* __restrict__ X,
                                                  const bf16* __restrict__ WT,
                                                  bf16* __restrict__ out, int trans) {
    const int lane = threadIdx.x & 31;
    const int lrow = lane & 15, lgrp = lane >> 4;
    const int mt = blockIdx.x;           // S/16 row tiles
    const int bh = blockIdx.y;
    const int b = bh >> 4, h = bh & 15;

    const bf16* A  = X + (size_t)b * S_ * D_;
    const bf16* BT = WT + (size_t)h * DH_ * D_;

    v8f c[4];
#pragma unroll
    for (int n = 0; n < 4; ++n) c[n] = (v8f){};

    // prologue loads
    v16bf a0 = load_a_frag(A, D_, mt * 16, 0, lrow, lgrp);
    v16bf b0[4];
#pragma unroll
    for (int n = 0; n < 4; ++n) b0[n] = load_b_frag(BT, D_, n * 16, 0, lrow, lgrp);

    for (int k = 0; k < D_ - 32; k += 32) {
        v16bf a1 = load_a_frag(A, D_, mt * 16, k + 32, lrow, lgrp);
        v16bf b1[4];
#pragma unroll
        for (int n = 0; n < 4; ++n)
            b1[n] = load_b_frag(BT, D_, n * 16, k + 32, lrow, lgrp);
#pragma unroll
        for (int n = 0; n < 4; ++n) c[n] = wmma_bf16(a0, b0[n], c[n]);
        a0 = a1;
#pragma unroll
        for (int n = 0; n < 4; ++n) b0[n] = b1[n];
    }
#pragma unroll
    for (int n = 0; n < 4; ++n) c[n] = wmma_bf16(a0, b0[n], c[n]);

    bf16* obase = out + (size_t)bh * S_ * DH_;
#pragma unroll
    for (int n = 0; n < 4; ++n) {
#pragma unroll
        for (int v = 0; v < 8; ++v) {
            int m  = mt * 16 + v + lgrp * 8;
            int nn = n * 16 + lrow;
            bf16 val = (bf16)c[n][v];
            if (trans) obase[(size_t)nn * S_ + m] = val;   // vT [DH][S]
            else       obase[(size_t)m * DH_ + nn] = val;  // qh/kh [S][DH]
        }
    }
}

// ---------------------------------------------------------------------------
// Flash attention: one wave handles 32 queries (two 16-row tiles) x full S for
// one (b,h). Every K/V fragment load is reused by both query tiles.
// ---------------------------------------------------------------------------
__global__ __launch_bounds__(32) void attn_kernel(const bf16* __restrict__ qh,
                                                  const bf16* __restrict__ kh,
                                                  const bf16* __restrict__ vT,
                                                  bf16* __restrict__ oatt) {
    const int lane = threadIdx.x & 31;
    const int lrow = lane & 15, lgrp = lane >> 4;
    const int qt0 = blockIdx.x * 2;            // two query tiles per wave
    const int bh  = blockIdx.y;
    const int b = bh >> 4, h = bh & 15;

    const bf16* qbase = qh + (size_t)bh * S_ * DH_;
    const bf16* kbase = kh + (size_t)bh * S_ * DH_;
    const bf16* vbase = vT + (size_t)bh * DH_ * S_;   // [DH][S]

    v16bf aq[2][2];
#pragma unroll
    for (int q = 0; q < 2; ++q) {
        aq[q][0] = load_a_frag(qbase, DH_, (qt0 + q) * 16, 0, lrow, lgrp);
        aq[q][1] = load_a_frag(qbase, DH_, (qt0 + q) * 16, 32, lrow, lgrp);
    }

    v8f acc[2][4];
    float mi[2][8], li[2][8];
#pragma unroll
    for (int q = 0; q < 2; ++q) {
#pragma unroll
        for (int n = 0; n < 4; ++n) acc[q][n] = (v8f){};
#pragma unroll
        for (int v = 0; v < 8; ++v) { mi[q][v] = -3.0e38f; li[q][v] = 0.f; }
    }

    __shared__ __align__(32) bf16 Pl[2][16 * 32];

    const float rscale = 0.125f;               // 1/sqrt(64)
    const int nblk = S_ / 32;

    for (int j = 0; j < nblk; ++j) {
        if (j + 1 < nblk) {
            __builtin_prefetch(kbase + (size_t)(j * 32 + 32) * DH_, 0, 0);
            __builtin_prefetch(vbase + (size_t)(j * 32 + 32), 0, 0);
        }

        v8f st[2][2];                          // [qtile][key subtile]
#pragma unroll
        for (int t = 0; t < 2; ++t) {
            int kb = j * 32 + t * 16;
            v16bf bk0 = load_b_frag(kbase, DH_, kb, 0, lrow, lgrp);
            v16bf bk1 = load_b_frag(kbase, DH_, kb, 32, lrow, lgrp);
#pragma unroll
            for (int q = 0; q < 2; ++q) {
                v8f c = {};
                c = wmma_bf16(aq[q][0], bk0, c);
                c = wmma_bf16(aq[q][1], bk1, c);
#pragma unroll
                for (int v = 0; v < 8; ++v) st[q][t][v] = c[v] * rscale;
            }
        }

        // Online softmax per query tile; row r lives in one 16-lane half.
#pragma unroll
        for (int q = 0; q < 2; ++q) {
#pragma unroll
            for (int v = 0; v < 8; ++v) {
                float mx = fmaxf(st[q][0][v], st[q][1][v]);
#pragma unroll
                for (int o = 8; o >= 1; o >>= 1) mx = fmaxf(mx, __shfl_xor(mx, o, 32));
                float mn    = fmaxf(mi[q][v], mx);
                float alpha = __expf(mi[q][v] - mn);
                float p0    = __expf(st[q][0][v] - mn);
                float p1    = __expf(st[q][1][v] - mn);
                float rs = p0 + p1;
#pragma unroll
                for (int o = 8; o >= 1; o >>= 1) rs += __shfl_xor(rs, o, 32);
                li[q][v] = li[q][v] * alpha + rs;
                mi[q][v] = mn;
#pragma unroll
                for (int n = 0; n < 4; ++n) acc[q][n][v] *= alpha;

                int m = v + lgrp * 8;          // restripe C-layout -> A-layout
                Pl[q][m * 32 + lrow]      = (bf16)p0;
                Pl[q][m * 32 + 16 + lrow] = (bf16)p1;
            }
        }
        __syncthreads();

        v16bf ap[2];
#pragma unroll
        for (int q = 0; q < 2; ++q)
            ap[q] = load_a_frag(&Pl[q][0], 32, 0, 0, lrow, lgrp);
#pragma unroll
        for (int n = 0; n < 4; ++n) {
            v16bf bv = load_b_frag(vbase, S_, n * 16, j * 32, lrow, lgrp);
#pragma unroll
            for (int q = 0; q < 2; ++q)
                acc[q][n] = wmma_bf16(ap[q], bv, acc[q][n]);
        }
        __syncthreads();
    }

    // Normalize and write concat-head layout [B][S][D] as bf16
#pragma unroll
    for (int q = 0; q < 2; ++q) {
#pragma unroll
        for (int v = 0; v < 8; ++v) {
            float inv = 1.0f / li[q][v];
            int m = (qt0 + q) * 16 + v + lgrp * 8;
#pragma unroll
            for (int n = 0; n < 4; ++n) {
                float o = acc[q][n][v] * inv;
                oatt[((size_t)b * S_ + m) * D_ + h * DH_ + n * 16 + lrow] = (bf16)o;
            }
        }
    }
}

// ---------------------------------------------------------------------------
// Output projection: 4-wave workgroups compute a 64x64 tile. The WoT k-slab
// (64 cols x 32 k, 4 KB) is double-buffered in LDS: the async global->LDS DMA
// for slab k+32 is issued BEFORE computing on slab k, so the ASYNCcnt-tracked
// copies overlap the WMMAs; drained at end of iteration before the barrier.
// ---------------------------------------------------------------------------
__global__ __launch_bounds__(128) void outproj_kernel(const bf16* __restrict__ A,
                                                      const bf16* __restrict__ WoT,
                                                      const float* __restrict__ bo,
                                                      float* __restrict__ out) {
    const int tid  = threadIdx.x;
    const int lane = tid & 31;
    const int wv   = tid >> 5;                 // wave 0..3 -> row tile
    const int lrow = lane & 15, lgrp = lane >> 4;
    const int mrow = (blockIdx.x * 4 + wv) * 16;
    const int nt0  = blockIdx.y * 4;           // 4 col tiles (64 cols)

    __shared__ __align__(16) bf16 Bt[2][64 * 32]; // double-buffered [n][k]

    auto stage = [&](int k, int buf) {
        // 256 16-byte chunks; 128 threads x 2 chunks.
        int ch = tid;
#pragma unroll
        for (int rep = 0; rep < 2; ++rep, ch += 128) {
            int n  = ch >> 2;                  // row 0..63
            int kk = (ch & 3) * 8;             // 8 bf16 = 16 bytes
            const bf16* g = WoT + (size_t)(nt0 * 16 + n) * D_ + k + kk;
            // low 32 bits of a generic LDS pointer == LDS byte offset
            unsigned ldsOff = (unsigned)(size_t)&Bt[buf][n * 32 + kk];
            asm volatile("global_load_async_to_lds_b128 %0, %1, off"
                         :: "v"(ldsOff), "v"(g) : "memory");
        }
    };

    v8f c[4];
#pragma unroll
    for (int n = 0; n < 4; ++n) c[n] = (v8f){};

    // prologue: stage slab 0
    stage(0, 0);
    asm volatile("s_wait_asynccnt 0x0" ::: "memory");
    __syncthreads();

    for (int k = 0; k < D_; k += 32) {
        const int buf = (k >> 5) & 1;
        if (k + 32 < D_) stage(k + 32, buf ^ 1);   // overlap DMA with compute

        v16bf a = load_a_frag(A, D_, mrow, k, lrow, lgrp);
#pragma unroll
        for (int n = 0; n < 4; ++n) {
            v16bf bb = load_b_frag(&Bt[buf][0], 32, n * 16, 0, lrow, lgrp);
            c[n] = wmma_bf16(a, bb, c[n]);
        }

        asm volatile("s_wait_asynccnt 0x0" ::: "memory");
        __syncthreads();
    }

#pragma unroll
    for (int n = 0; n < 4; ++n) {
        int nn = nt0 * 16 + n * 16 + lrow;
        float bias = bo[nn];
#pragma unroll
        for (int v = 0; v < 8; ++v) {
            int m = mrow + v + lgrp * 8;
            out[(size_t)m * D_ + nn] = c[n][v] + bias;
        }
    }
}

// ---------------------------------------------------------------------------
extern "C" void kernel_launch(void* const* d_in, const int* in_sizes, int n_in,
                              void* d_out, int out_size, void* d_ws, size_t ws_size,
                              hipStream_t stream) {
    const float* Q  = (const float*)d_in[0];
    const float* K  = (const float*)d_in[1];
    const float* V  = (const float*)d_in[2];
    const float* Wq = (const float*)d_in[3];
    const float* Wk = (const float*)d_in[4];
    const float* Wv = (const float*)d_in[5];
    const float* Wo = (const float*)d_in[6];
    const float* bo = (const float*)d_in[7];
    float* out = (float*)d_out;

    // Workspace carve-up (~67 MB total), 256B aligned
    char* base = (char*)d_ws;
    size_t off = 0;
    auto carve = [&](size_t bytes) -> char* {
        char* r = base + off;
        off = (off + bytes + 255) & ~(size_t)255;
        return r;
    };
    const size_t szQKV = (size_t)B_ * S_ * D_ * sizeof(bf16);
    const size_t szW   = (size_t)H_ * D_ * DH_ * sizeof(bf16);
    const size_t szWo  = (size_t)D_ * D_ * sizeof(bf16);
    const size_t szHd  = (size_t)B_ * H_ * S_ * DH_ * sizeof(bf16);

    bf16* Qb   = (bf16*)carve(szQKV);
    bf16* Kb   = (bf16*)carve(szQKV);
    bf16* Vb   = (bf16*)carve(szQKV);
    bf16* WqT  = (bf16*)carve(szW);
    bf16* WkT  = (bf16*)carve(szW);
    bf16* WvT  = (bf16*)carve(szW);
    bf16* WoT  = (bf16*)carve(szWo);
    bf16* qhB  = (bf16*)carve(szHd);
    bf16* khB  = (bf16*)carve(szHd);
    bf16* vTr  = (bf16*)carve(szHd);
    bf16* oatt = (bf16*)carve(szQKV);

    // 1) Precision conversion + weight transposes
    const int nQ = B_ * S_ * D_;
    cvt_f32_bf16<<<2048, 256, 0, stream>>>(Q, Qb, nQ);
    cvt_f32_bf16<<<2048, 256, 0, stream>>>(K, Kb, nQ);
    cvt_f32_bf16<<<2048, 256, 0, stream>>>(V, Vb, nQ);
    cvt_w_trans<<<(H_ * D_ * DH_ + 255) / 256, 256, 0, stream>>>(Wq, WqT);
    cvt_w_trans<<<(H_ * D_ * DH_ + 255) / 256, 256, 0, stream>>>(Wk, WkT);
    cvt_w_trans<<<(H_ * D_ * DH_ + 255) / 256, 256, 0, stream>>>(Wv, WvT);
    cvt_wo_trans<<<(D_ * D_ + 255) / 256, 256, 0, stream>>>(Wo, WoT);

    // 2) Per-head projections (V written transposed for the PV GEMM)
    dim3 pg(S_ / 16, B_ * H_);
    proj_kernel<<<pg, 32, 0, stream>>>(Qb, WqT, qhB, 0);
    proj_kernel<<<pg, 32, 0, stream>>>(Kb, WkT, khB, 0);
    proj_kernel<<<pg, 32, 0, stream>>>(Vb, WvT, vTr, 1);

    // 3) Flash attention (one wave per 32 queries per head)
    attn_kernel<<<dim3(S_ / 32, B_ * H_), 32, 0, stream>>>(qhB, khB, vTr, oatt);

    // 4) Output projection + bias (4-wave blocks, double-buffered async LDS)
    outproj_kernel<<<dim3(B_ * S_ / 64, D_ / 64), 128, 0, stream>>>(oatt, WoT, bo, out);
}
